// FANBlock_SE_29111288333049
// MI455X (gfx1250) — matrix operand order repaired and checked by reading, not compiled, not run on hardware
//
#include <hip/hip_runtime.h>

// ---------------------------------------------------------------------------
// FANBlock+SE for MI455X (gfx1250, wave32, WMMA).
// GEMMs + attention on v_wmma_f32_16x16x32_bf16; GEMM A-tiles staged with
// global_load_async_to_lds_b128 (ASYNCcnt) and double-buffered LDS.
// ---------------------------------------------------------------------------

typedef __attribute__((ext_vector_type(16))) __bf16 v16bf;
typedef __attribute__((ext_vector_type(8)))  float  v8f;
typedef unsigned short u16;
typedef unsigned int   u32;

union Frag { u32 u[8]; v16bf v; };

__device__ __forceinline__ u16 f2bf(float f) {
  u32 x = __float_as_uint(f);
  u32 r = x + 0x7FFFu + ((x >> 16) & 1u);   // round-to-nearest-even
  return (u16)(r >> 16);
}
__device__ __forceinline__ float bf2f(u16 u) {
  return __uint_as_float(((u32)u) << 16);
}

// CDNA5 async global->LDS copy (16B per lane), tracked with ASYNCcnt.
__device__ __forceinline__ void async_load_b128(u32 lds_off, const void* gptr) {
  asm volatile("global_load_async_to_lds_b128 %0, %1, off"
               :: "v"(lds_off), "v"((unsigned long long)(uintptr_t)gptr)
               : "memory");
}
__device__ __forceinline__ void wait_async0() { asm volatile("s_wait_asynccnt 0x0" ::: "memory"); }
__device__ __forceinline__ void wait_async2() { asm volatile("s_wait_asynccnt 0x2" ::: "memory"); }

#define Mrows 12544      // B*N = 16*784
#define Cdim  384
#define HIDd  1536
#define Ntok  784

// ---------------------- fp32 -> bf16 convert (weights) ----------------------
__global__ void k_f2bf(const float* __restrict__ in, u16* __restrict__ out, int n) {
  int i = blockIdx.x * 256 + threadIdx.x;
  if (i < n) out[i] = f2bf(in[i]);
}

// ---------------------- LayerNorm rows of [M, C], bf16 out ------------------
__global__ __launch_bounds__(128) void k_ln(const float* __restrict__ x,
                                            const float* __restrict__ g,
                                            const float* __restrict__ b,
                                            u16* __restrict__ out, int C) {
  __shared__ float red[128];
  int row = blockIdx.x, t = threadIdx.x;
  const float* xr = x + (size_t)row * C;
  float s = 0.f, s2 = 0.f;
  for (int c = t; c < C; c += 128) { float v = xr[c]; s += v; s2 += v * v; }
  red[t] = s; __syncthreads();
  for (int o = 64; o > 0; o >>= 1) { if (t < o) red[t] += red[t + o]; __syncthreads(); }
  float mean = red[0] / (float)C; __syncthreads();
  red[t] = s2; __syncthreads();
  for (int o = 64; o > 0; o >>= 1) { if (t < o) red[t] += red[t + o]; __syncthreads(); }
  float var  = red[0] / (float)C - mean * mean;
  float rstd = rsqrtf(var + 1e-5f);
  for (int c = t; c < C; c += 128)
    out[(size_t)row * C + c] = f2bf((xr[c] - mean) * rstd * g[c] + b[c]);
}

// ---------------------- WMMA bf16 GEMM: out = A[MxK] @ B[KxN] + bias --------
// Tile 128x64, K-step 32, double-buffered LDS. A staged via async B128 DMA,
// B staged via one uint4 load/thread transposed into LDS. 8 waves; each wave
// owns a 32x32 subtile = 2x2 v_wmma_f32_16x16x32_bf16 fragments.
__global__ __launch_bounds__(256) void k_gemm(const u16* __restrict__ A,
                                              const u16* __restrict__ Bm,
                                              const float* __restrict__ bias,
                                              float* __restrict__ outF,
                                              u16* __restrict__ outB,
                                              int M, int N, int K) {
  __shared__ u16 As[2][128 * 40];   // [row][k], stride 40 halves (80B, 16B-aligned)
  __shared__ u16 BsT[2][64 * 34];   // [n][k] transposed, stride 34
  const int t = threadIdx.x;
  const int lane = t & 31, w = t >> 5;
  const int hl = lane >> 4, ll = lane & 15;
  const int wm = (w & 3) * 32, wn = (w >> 2) * 32;
  const int m0 = blockIdx.x * 128, n0 = blockIdx.y * 64;
  const int KT = K >> 5;

  // staging index precompute
  const int ar0 = t >> 2,        ac0 = (t & 3) * 8;         // A chunk 0: idx=t
  const int ar1 = (t + 256) >> 2, ac1 = ((t + 256) & 3) * 8; // A chunk 1: idx=t+256
  const int bkr = t >> 3, bc8 = (t & 7) * 8;                 // B chunk

  auto stage = [&](int kt, int buf) {
    const int k0 = kt * 32;
    // A tile: 128x32 halves via 2 async B128 copies per thread
    async_load_b128((u32)(uintptr_t)&As[buf][ar0 * 40 + ac0],
                    &A[(size_t)(m0 + ar0) * K + k0 + ac0]);
    async_load_b128((u32)(uintptr_t)&As[buf][ar1 * 40 + ac1],
                    &A[(size_t)(m0 + ar1) * K + k0 + ac1]);
    // B tile: 32x64 halves, one uint4 per thread, transposed into LDS
    uint4 wv = *(const uint4*)&Bm[(size_t)(k0 + bkr) * N + n0 + bc8];
    u16* bs = &BsT[buf][0];
    bs[(bc8 + 0) * 34 + bkr] = (u16)(wv.x & 0xFFFFu);
    bs[(bc8 + 1) * 34 + bkr] = (u16)(wv.x >> 16);
    bs[(bc8 + 2) * 34 + bkr] = (u16)(wv.y & 0xFFFFu);
    bs[(bc8 + 3) * 34 + bkr] = (u16)(wv.y >> 16);
    bs[(bc8 + 4) * 34 + bkr] = (u16)(wv.z & 0xFFFFu);
    bs[(bc8 + 5) * 34 + bkr] = (u16)(wv.z >> 16);
    bs[(bc8 + 6) * 34 + bkr] = (u16)(wv.w & 0xFFFFu);
    bs[(bc8 + 7) * 34 + bkr] = (u16)(wv.w >> 16);
  };

  v8f acc[2][2] = {};
  stage(0, 0);
  for (int kt = 0; kt < KT; ++kt) {
    const int cur = kt & 1;
    if (kt + 1 < KT) { stage(kt + 1, cur ^ 1); wait_async2(); }
    else             { wait_async0(); }
    __syncthreads();
    Frag a[2], b[2];
    for (int fm = 0; fm < 2; ++fm) {
      const u16* ap = &As[cur][(wm + fm * 16 + ll) * 40 + hl * 16];
      for (int j = 0; j < 4; ++j) {
        a[fm].u[j]     = *(const u32*)&ap[2 * j];
        a[fm].u[4 + j] = *(const u32*)&ap[8 + 2 * j];
      }
    }
    for (int fn = 0; fn < 2; ++fn) {
      const u16* bp = &BsT[cur][(wn + fn * 16 + ll) * 34 + hl * 16];
      for (int j = 0; j < 8; ++j) b[fn].u[j] = *(const u32*)&bp[2 * j];
    }
    for (int fm = 0; fm < 2; ++fm)
      for (int fn = 0; fn < 2; ++fn)
        acc[fm][fn] = __builtin_amdgcn_wmma_f32_16x16x32_bf16(
            false, a[fm].v, false, b[fn].v, (short)0, acc[fm][fn], false, false);
    __syncthreads();   // protect buffer reuse by stage(kt+2)
  }

  if (outF) {
    for (int fm = 0; fm < 2; ++fm)
      for (int fn = 0; fn < 2; ++fn) {
        int col = n0 + wn + fn * 16 + ll;
        float bv = bias[col];
        for (int r = 0; r < 8; ++r) {
          int row = m0 + wm + fm * 16 + r + 8 * hl;
          outF[(size_t)row * N + col] = acc[fm][fn][r] + bv;
        }
      }
  }
  if (outB) {
    for (int fm = 0; fm < 2; ++fm)
      for (int fn = 0; fn < 2; ++fn) {
        int col = n0 + wn + fn * 16 + ll;
        float bv = bias[col];
        for (int r = 0; r < 8; ++r) {
          int row = m0 + wm + fm * 16 + r + 8 * hl;
          outB[(size_t)row * N + col] = f2bf(acc[fm][fn][r] + bv);
        }
      }
  }
}

// ---------------------- Flash attention (d=48 padded to 64) -----------------
// grid.x = B * HEADS * 13 query tiles of 64. 128 threads = 4 waves;
// wave w owns 16 query rows (w*16 .. w*16+15).
__global__ __launch_bounds__(128) void k_attn(const u16* __restrict__ Q,
                                              const u16* __restrict__ KV,
                                              u16* __restrict__ O) {
  __shared__ u16 Qs[64 * 68];
  __shared__ u16 Ks[64 * 68];
  __shared__ u16 VsT[64 * 68];   // [d][key]
  __shared__ u16 Ps[64 * 68];    // wave-private 16-row slabs
  int bx = blockIdx.x;
  int qt = bx % 13, h = (bx / 13) & 7, b = bx / (13 * 8);
  int t = threadIdx.x, lane = t & 31, w = t >> 5;
  int hl = lane >> 4, ll = lane & 15;
  const float scale = 0.14433756729740643f;    // 48^-0.5
  for (int i = 0; i < 32; ++i) {
    int e = i * 128 + t, row = e >> 6, d = e & 63;
    int q = qt * 64 + row;
    u16 v = 0;
    if (q < Ntok && d < 48) v = Q[((size_t)(b * Ntok + q)) * Cdim + h * 48 + d];
    Qs[row * 68 + d] = v;
  }
  float m_r[8], l_r[8];
  v8f o_acc[3] = {};
  for (int r = 0; r < 8; ++r) { m_r[r] = -1e30f; l_r[r] = 0.f; }
  for (int kc = 0; kc < 13; ++kc) {
    __syncthreads();
    for (int i = 0; i < 32; ++i) {
      int e = i * 128 + t, row = e >> 6, d = e & 63;
      int key = kc * 64 + row;
      u16 kvv = 0, vvv = 0;
      if (key < Ntok && d < 48) {
        size_t base = ((size_t)(b * Ntok + key)) * (2 * Cdim) + h * 48 + d;
        kvv = KV[base];
        vvv = KV[base + Cdim];
      }
      Ks[row * 68 + d]  = kvv;
      VsT[d * 68 + row] = vvv;
    }
    __syncthreads();
    // S = Q K^T
    v8f s_acc[4] = {};
    for (int ks = 0; ks < 2; ++ks) {
      Frag a;
      const u16* ap = &Qs[(w * 16 + ll) * 68 + ks * 32 + hl * 16];
      for (int j = 0; j < 4; ++j) {
        a.u[j]     = *(const u32*)&ap[2 * j];
        a.u[4 + j] = *(const u32*)&ap[8 + 2 * j];
      }
      for (int f = 0; f < 4; ++f) {
        Frag bf;
        const u16* bp = &Ks[(f * 16 + ll) * 68 + ks * 32 + hl * 16];
        for (int j = 0; j < 8; ++j) bf.u[j] = *(const u32*)&bp[2 * j];
        s_acc[f] = __builtin_amdgcn_wmma_f32_16x16x32_bf16(
            false, a.v, false, bf.v, (short)0, s_acc[f], false, false);
      }
    }
    // online softmax
    float sv[4][8];
    for (int f = 0; f < 4; ++f) {
      int key = kc * 64 + f * 16 + ll;
      bool valid = key < Ntok;
      for (int r = 0; r < 8; ++r) sv[f][r] = valid ? s_acc[f][r] * scale : -1e30f;
    }
    float mnew[8], fr[8], rs[8];
    for (int r = 0; r < 8; ++r) {
      float mx = fmaxf(fmaxf(sv[0][r], sv[1][r]), fmaxf(sv[2][r], sv[3][r]));
      for (int o = 8; o >= 1; o >>= 1) mx = fmaxf(mx, __shfl_xor(mx, o, 32));
      mnew[r] = fmaxf(m_r[r], mx);
      fr[r] = __expf(m_r[r] - mnew[r]);
      rs[r] = 0.f;
    }
    for (int f = 0; f < 4; ++f)
      for (int r = 0; r < 8; ++r) {
        float p = __expf(sv[f][r] - mnew[r]);
        rs[r] += p;
        Ps[(w * 16 + r + 8 * hl) * 68 + f * 16 + ll] = f2bf(p);
      }
    for (int r = 0; r < 8; ++r) {
      float x = rs[r];
      for (int o = 8; o >= 1; o >>= 1) x += __shfl_xor(x, o, 32);
      l_r[r] = l_r[r] * fr[r] + x;
      m_r[r] = mnew[r];
    }
    for (int f = 0; f < 3; ++f)
      for (int r = 0; r < 8; ++r) o_acc[f][r] = o_acc[f][r] * fr[r];
    __syncthreads();
    // O += P V (d-frag 3 is padding)
    for (int ks = 0; ks < 2; ++ks) {
      Frag a;
      const u16* ap = &Ps[(w * 16 + ll) * 68 + ks * 32 + hl * 16];
      for (int j = 0; j < 4; ++j) {
        a.u[j]     = *(const u32*)&ap[2 * j];
        a.u[4 + j] = *(const u32*)&ap[8 + 2 * j];
      }
      for (int f = 0; f < 3; ++f) {
        Frag bf;
        const u16* bp = &VsT[(f * 16 + ll) * 68 + ks * 32 + hl * 16];
        for (int j = 0; j < 8; ++j) bf.u[j] = *(const u32*)&bp[2 * j];
        o_acc[f] = __builtin_amdgcn_wmma_f32_16x16x32_bf16(
            false, a.v, false, bf.v, (short)0, o_acc[f], false, false);
      }
    }
  }
  for (int f = 0; f < 3; ++f)
    for (int r = 0; r < 8; ++r) {
      int q = qt * 64 + w * 16 + r + 8 * hl;
      if (q < Ntok) {
        int d = f * 16 + ll;
        O[((size_t)(b * Ntok + q)) * Cdim + h * 48 + d] = f2bf(o_acc[f][r] / l_r[r]);
      }
    }
}

// ---------------------- x1 = x + gamma1 * proj ------------------------------
__global__ void k_resid1(const float* __restrict__ x, const float* __restrict__ p,
                         const float* __restrict__ g1, float* __restrict__ x1, int total) {
  int i = blockIdx.x * 256 + threadIdx.x;
  if (i >= total) return;
  x1[i] = x[i] + g1[i % Cdim] * p[i];
}

// ---------------------- depthwise 3x3 + mlp_gamma residual -> bf16 ----------
__global__ void k_dwconv(const float* __restrict__ hf, const float* __restrict__ dww,
                         const float* __restrict__ dwb, const float* __restrict__ mg,
                         u16* __restrict__ out, int total) {
  int idx = blockIdx.x * 256 + threadIdx.x;
  if (idx >= total) return;
  int c = idx % HIDd;
  int n = (idx / HIDd) % Ntok;
  int bb = idx / (HIDd * Ntok);
  int y = n / 28, x = n % 28;
  float acc = dwb[c];
  for (int dy = -1; dy <= 1; ++dy) {
    int yy = y + dy; if (yy < 0 || yy >= 28) continue;
    for (int dx = -1; dx <= 1; ++dx) {
      int xx = x + dx; if (xx < 0 || xx >= 28) continue;
      acc += hf[((size_t)(bb * Ntok + yy * 28 + xx)) * HIDd + c] * dww[c * 9 + (dy + 1) * 3 + (dx + 1)];
    }
  }
  out[idx] = f2bf(mg[c] * acc + hf[idx]);
}

// ---------------------- SE squeeze: mean over spatial -----------------------
__global__ void k_semean(const float* __restrict__ h2, float* __restrict__ s) {
  int idx = blockIdx.x * 256 + threadIdx.x;
  if (idx >= 16 * Cdim) return;
  int bb = idx / Cdim, c = idx % Cdim;
  float acc = 0.f;
  for (int n = 0; n < Ntok; ++n) acc += h2[((size_t)(bb * Ntok + n)) * Cdim + c];
  s[idx] = acc * (1.0f / (float)Ntok);
}

// ---------------------- SE MLP: relu -> sigmoid -----------------------------
__global__ __launch_bounds__(128) void k_semlp(const float* __restrict__ s,
                                               const float* __restrict__ rw,
                                               const float* __restrict__ rb,
                                               const float* __restrict__ ew,
                                               const float* __restrict__ eb,
                                               float* __restrict__ g) {
  __shared__ float sh_s[Cdim];
  __shared__ float sh_t[96];
  int bb = blockIdx.x, t = threadIdx.x;
  for (int c = t; c < Cdim; c += 128) sh_s[c] = s[bb * Cdim + c];
  __syncthreads();
  for (int r = t; r < 96; r += 128) {
    float acc = rb[r];
    for (int c = 0; c < Cdim; ++c) acc += sh_s[c] * rw[c * 96 + r];
    sh_t[r] = fmaxf(acc, 0.f);
  }
  __syncthreads();
  for (int c = t; c < Cdim; c += 128) {
    float acc = eb[c];
    for (int r = 0; r < 96; ++r) acc += sh_t[r] * ew[r * Cdim + c];
    g[bb * Cdim + c] = 1.f / (1.f + __expf(-acc));
  }
}

// ---------------------- out = x1 + gamma2 * (h2 * gate) ---------------------
__global__ void k_final(const float* __restrict__ x1, const float* __restrict__ h2,
                        const float* __restrict__ g2, const float* __restrict__ gate,
                        float* __restrict__ out, int total) {
  int i = blockIdx.x * 256 + threadIdx.x;
  if (i >= total) return;
  int c = i % Cdim;
  int bb = i / (Ntok * Cdim);
  out[i] = x1[i] + g2[c] * h2[i] * gate[bb * Cdim + c];
}

// ---------------------------------------------------------------------------
extern "C" void kernel_launch(void* const* d_in, const int* in_sizes, int n_in,
                              void* d_out, int out_size, void* d_ws, size_t ws_size,
                              hipStream_t stream) {
  const float* x      = (const float*)d_in[0];
  const float* n1g    = (const float*)d_in[3];
  const float* n1b    = (const float*)d_in[4];
  const float* q_w    = (const float*)d_in[5];
  const float* q_b    = (const float*)d_in[6];
  const float* kv_w   = (const float*)d_in[7];
  const float* kv_b   = (const float*)d_in[8];
  const float* proj_w = (const float*)d_in[9];
  const float* proj_b = (const float*)d_in[10];
  const float* gamma1 = (const float*)d_in[11];
  const float* n2g    = (const float*)d_in[12];
  const float* n2b    = (const float*)d_in[13];
  const float* fc1_w  = (const float*)d_in[14];
  const float* fc1_b  = (const float*)d_in[15];
  const float* dw_w   = (const float*)d_in[16];
  const float* dw_b   = (const float*)d_in[17];
  const float* mlp_g  = (const float*)d_in[18];
  const float* fc2_w  = (const float*)d_in[19];
  const float* fc2_b  = (const float*)d_in[20];
  const float* se_rw  = (const float*)d_in[21];
  const float* se_rb  = (const float*)d_in[22];
  const float* se_ew  = (const float*)d_in[23];
  const float* se_eb  = (const float*)d_in[24];
  const float* gamma2 = (const float*)d_in[25];
  float* out = (float*)d_out;

  char* p = (char*)d_ws;
  auto carve = [&](size_t bytes) { char* r = p; p += (bytes + 255) & ~(size_t)255; return (void*)r; };
  u16*   wq    = (u16*)  carve((size_t)Cdim * Cdim * 2);
  u16*   wkv   = (u16*)  carve((size_t)Cdim * 2 * Cdim * 2);
  u16*   wproj = (u16*)  carve((size_t)Cdim * Cdim * 2);
  u16*   wfc1  = (u16*)  carve((size_t)Cdim * HIDd * 2);
  u16*   wfc2  = (u16*)  carve((size_t)HIDd * Cdim * 2);
  u16*   h1b   = (u16*)  carve((size_t)Mrows * Cdim * 2);
  u16*   qb    = (u16*)  carve((size_t)Mrows * Cdim * 2);
  u16*   kvb   = (u16*)  carve((size_t)Mrows * 2 * Cdim * 2);
  u16*   ob    = (u16*)  carve((size_t)Mrows * Cdim * 2);
  float* projf = (float*)carve((size_t)Mrows * Cdim * 4);   // reused for fc2 out
  float* x1    = (float*)carve((size_t)Mrows * Cdim * 4);
  float* hfc1  = (float*)carve((size_t)Mrows * HIDd * 4);
  u16*   hmb   = (u16*)  carve((size_t)Mrows * HIDd * 2);
  float* sbuf  = (float*)carve((size_t)16 * Cdim * 4);
  float* gbuf  = (float*)carve((size_t)16 * Cdim * 4);
  float* h2    = projf;

  const int totC = Mrows * Cdim;
  const int totH = Mrows * HIDd;

  k_f2bf<<<(Cdim*Cdim + 255)/256, 256, 0, stream>>>(q_w,    wq,    Cdim*Cdim);
  k_f2bf<<<(Cdim*2*Cdim + 255)/256, 256, 0, stream>>>(kv_w, wkv,   Cdim*2*Cdim);
  k_f2bf<<<(Cdim*Cdim + 255)/256, 256, 0, stream>>>(proj_w, wproj, Cdim*Cdim);
  k_f2bf<<<(Cdim*HIDd + 255)/256, 256, 0, stream>>>(fc1_w,  wfc1,  Cdim*HIDd);
  k_f2bf<<<(HIDd*Cdim + 255)/256, 256, 0, stream>>>(fc2_w,  wfc2,  HIDd*Cdim);

  k_ln<<<Mrows, 128, 0, stream>>>(x, n1g, n1b, h1b, Cdim);
  k_gemm<<<dim3(Mrows/128, Cdim/64),   256, 0, stream>>>(h1b, wq,  q_b,  nullptr, qb,  Mrows, Cdim,   Cdim);
  k_gemm<<<dim3(Mrows/128, 2*Cdim/64), 256, 0, stream>>>(h1b, wkv, kv_b, nullptr, kvb, Mrows, 2*Cdim, Cdim);
  k_attn<<<16 * 8 * 13, 128, 0, stream>>>(qb, kvb, ob);
  k_gemm<<<dim3(Mrows/128, Cdim/64), 256, 0, stream>>>(ob, wproj, proj_b, projf, nullptr, Mrows, Cdim, Cdim);
  k_resid1<<<(totC + 255)/256, 256, 0, stream>>>(x, projf, gamma1, x1, totC);
  k_ln<<<Mrows, 128, 0, stream>>>(x1, n2g, n2b, h1b, Cdim);
  k_gemm<<<dim3(Mrows/128, HIDd/64), 256, 0, stream>>>(h1b, wfc1, fc1_b, hfc1, nullptr, Mrows, HIDd, Cdim);
  k_dwconv<<<(totH + 255)/256, 256, 0, stream>>>(hfc1, dw_w, dw_b, mlp_g, hmb, totH);
  k_gemm<<<dim3(Mrows/128, Cdim/64), 256, 0, stream>>>(hmb, wfc2, fc2_b, h2, nullptr, Mrows, Cdim, HIDd);
  k_semean<<<(16*Cdim + 255)/256, 256, 0, stream>>>(h2, sbuf);
  k_semlp<<<16, 128, 0, stream>>>(sbuf, se_rw, se_rb, se_ew, se_eb, gbuf);
  k_final<<<(totC + 255)/256, 256, 0, stream>>>(x1, h2, gamma2, gbuf, out, totC);
}
